// SymmetricFaceConv_3951369912809
// MI455X (gfx1250) — matrix operand order repaired and checked by reading, not compiled, not run on hardware
//
#include <hip/hip_runtime.h>
#include <hip/hip_bf16.h>

typedef __attribute__((ext_vector_type(16))) __bf16 v16bf;
typedef __attribute__((ext_vector_type(2)))  __bf16 v2bf;
typedef __attribute__((ext_vector_type(8)))  float  v8f;

#define N_REAL   262144
#define C_IN     128
#define C_OUT    128
#define KNB      9
#define TILE_M   64
#define THREADS  256
// LDS row stride in bf16 elements: 134 (= 67 dwords). 67 mod 64 = 3 is odd and
// 3*d != 4,8 (mod 64) for |d|<=15  => A- and B-fragment DS reads are bank-conflict free.
#define WSTRIDE  134

// round-to-nearest-even f32 -> bf16 using only __bf16 storage
__device__ __forceinline__ __bf16 f2bf(float f) {
    union { float f; unsigned u; } v; v.f = f;
    unsigned r = (v.u + 0x7FFFu + ((v.u >> 16) & 1u)) >> 16;
    union { unsigned short s; __bf16 b; } o; o.s = (unsigned short)r;
    return o.b;
}

__global__ __launch_bounds__(THREADS)
void SymmetricFaceConv_kernel(const float* __restrict__ x,
                              const float* __restrict__ w0,
                              const float* __restrict__ w1,
                              const float* __restrict__ w2,
                              const float* __restrict__ bias,
                              const long long* __restrict__ nbr,
                              float* __restrict__ out)
{
    // Weights: (o, c) row-major in bf16; padded rows for bank-conflict-free DS reads.
    __shared__ __bf16 Wl[3][C_OUT][WSTRIDE];   // 102912 B
    __shared__ __bf16 Al[3][TILE_M][WSTRIDE];  //  51456 B
    __shared__ float  biasl[C_OUT];            //    512 B

    const int tid   = threadIdx.x;
    const int mbase = blockIdx.x * TILE_M;

    // ---- stage weights + bias into LDS (bf16) ----
    {
        const float* ws[3] = { w0, w1, w2 };
        #pragma unroll
        for (int s = 0; s < 3; ++s) {
            const float* w = ws[s];
            for (int i = tid; i < C_OUT * C_IN; i += THREADS) {
                int o = i >> 7, c = i & (C_IN - 1);
                Wl[s][o][c] = f2bf(w[i]);
            }
        }
        if (tid < C_OUT) biasl[tid] = bias[tid];
    }

    // ---- gather + symmetric neighbor sums (exact in f32), round once to bf16 ----
    {
        const int c  = tid & (C_IN - 1);
        const int rh = tid >> 7;                 // 0 or 1: two rows in flight
        for (int ro = 0; ro < TILE_M / 2; ++ro) {
            const int r = ro * 2 + rh;
            const long long* nr = nbr + (long long)(mbase + r) * KNB;
            long long i0 = nr[0];
            float s0 = (i0 < N_REAL) ? x[i0 * C_IN + c] : 0.0f;
            float s1 = 0.0f, s2 = 0.0f;
            #pragma unroll
            for (int k = 1; k < KNB; k += 2) {
                long long i = nr[k];
                if (i < N_REAL) s1 += x[i * C_IN + c];
            }
            #pragma unroll
            for (int k = 2; k < KNB; k += 2) {
                long long i = nr[k];
                if (i < N_REAL) s2 += x[i * C_IN + c];
            }
            Al[0][r][c] = f2bf(s0);
            Al[1][r][c] = f2bf(s1);
            Al[2][r][c] = f2bf(s2);
        }
    }
    __syncthreads();

    // ---- WMMA: out(64x128) = sum_s A_s(64x128) * W_s^T(128x128) + bias ----
    const int wave = tid >> 5;
    const int lane = tid & 31;
    const int half = lane >> 4;        // lane half selects K (A/B) or M+8 (C/D)
    const int l16  = lane & 15;
    const int nt   = wave;             // this wave's 16-wide output column tile
    const int ob   = nt * 16 + l16;    // output column (= weight row o) for this lane

    const float bv = biasl[ob];
    v8f acc[4];
    #pragma unroll
    for (int mt = 0; mt < 4; ++mt)
        #pragma unroll
        for (int d = 0; d < 8; ++d) acc[mt][d] = bv;

    #pragma unroll
    for (int s = 0; s < 3; ++s) {
        #pragma unroll
        for (int kb = 0; kb < 4; ++kb) {
            const int k0 = kb * 32;
            // B fragment (32x16, K x N): B[k][n] = W[n][k]; lane holds column n = ob.
            // VGPR v<4: K = k0+2v(+8 for hi lanes); v>=4: K = k0+2v+8(+8), pairs +0,+1.
            v16bf bf;
            #pragma unroll
            for (int v = 0; v < 8; ++v) {
                const int kk = k0 + ((v < 4) ? 2 * v : 2 * v + 8) + (half ? 8 : 0);
                v2bf p = *(const v2bf*)&Wl[s][ob][kk];
                bf[2 * v]     = p[0];
                bf[2 * v + 1] = p[1];
            }
            #pragma unroll
            for (int mt = 0; mt < 4; ++mt) {
                const int row = mt * 16 + l16;   // A rows: M = lane%16 for both halves
                // A fragment (16x32): VGPR v<4: K = 2v(+8 for hi lanes); v>=4: K = 2v+8(+8).
                v16bf af;
                #pragma unroll
                for (int v = 0; v < 8; ++v) {
                    const int kk = k0 + ((v < 4) ? 2 * v : 2 * v + 8) + (half ? 8 : 0);
                    v2bf p = *(const v2bf*)&Al[s][row][kk];
                    af[2 * v]     = p[0];
                    af[2 * v + 1] = p[1];
                }
                acc[mt] = __builtin_amdgcn_wmma_f32_16x16x32_bf16(
                    /*neg_a=*/false, af, /*neg_b=*/false, bf,
                    /*c_mod=*/(short)0, acc[mt],
                    /*reuse_a=*/false, /*reuse_b=*/false);
            }
        }
    }

    // ---- store D: VGPR d -> M = d (lanes0-15) / d+8 (lanes16-31); N = lane%16 ----
    #pragma unroll
    for (int mt = 0; mt < 4; ++mt) {
        #pragma unroll
        for (int d = 0; d < 8; ++d) {
            const int row = mbase + mt * 16 + d + half * 8;
            out[(long long)row * C_OUT + ob] = acc[mt][d];
        }
    }
}

extern "C" void kernel_launch(void* const* d_in, const int* in_sizes, int n_in,
                              void* d_out, int out_size, void* d_ws, size_t ws_size,
                              hipStream_t stream) {
    const float*     x    = (const float*)d_in[0];
    const float*     w0   = (const float*)d_in[1];
    const float*     w1   = (const float*)d_in[2];
    const float*     w2   = (const float*)d_in[3];
    const float*     bias = (const float*)d_in[4];
    const long long* nbr  = (const long long*)d_in[5];
    // d_in[6] = face_is_pad (unused: pads are exactly indices >= N_REAL)
    // d_in[7] = pad_size    (unused: compile-time constant)
    float* out = (float*)d_out;

    const int grid = N_REAL / TILE_M;  // 4096 blocks of 64 rows
    SymmetricFaceConv_kernel<<<grid, THREADS, 0, stream>>>(x, w0, w1, w2, bias, nbr, out);
}
